// KoLeoLoss_20547123544251
// MI455X (gfx1250) — matrix operand order repaired and checked by reading, not compiled, not run on hardware
//
#include <hip/hip_runtime.h>
#include <math.h>

#define N_ROWS 16384
#define DIMS   256

typedef __attribute__((ext_vector_type(16))) __bf16        v16bf;
typedef __attribute__((ext_vector_type(8)))  float         v8f;
typedef __attribute__((ext_vector_type(4)))  unsigned int  u32x4;
typedef __attribute__((ext_vector_type(4)))  float         f32x4;

union Frag { v16bf v; u32x4 q[2]; };

__device__ __forceinline__ unsigned short f2bf(float f) {
    unsigned int u = __float_as_uint(f);
    u += 0x7FFFu + ((u >> 16) & 1u);           // round-to-nearest-even
    return (unsigned short)(u >> 16);
}
__device__ __forceinline__ float bf2f(unsigned short h) {
    return __uint_as_float(((unsigned int)h) << 16);
}

// Async DMA: global (SGPR base + VGPR byte offset) -> LDS, 16 bytes, ASYNCcnt.
__device__ __forceinline__ void async_ld16(unsigned lds_addr, const void* base,
                                           unsigned voff) {
    asm volatile("global_load_async_to_lds_b128 %0, %1, %2"
                 :: "v"(lds_addr), "v"(voff), "s"(base)
                 : "memory");
}
__device__ __forceinline__ void wait_async0() {
    asm volatile("s_wait_asynccnt 0" ::: "memory");
}

// ---------------------------------------------------------------------------
// Kernel 1: L2-normalize rows (eps=1e-12) and split to bf16 hi + lo residual.
// One wave (32 lanes) per row; 8 fp32 per lane.
// ---------------------------------------------------------------------------
__global__ void __launch_bounds__(256) koleo_normalize_split(
    const float* __restrict__ x,
    unsigned short* __restrict__ ghi,
    unsigned short* __restrict__ glo)
{
    const int wave = threadIdx.x >> 5;
    const int lane = threadIdx.x & 31;
    const int row  = blockIdx.x * 8 + wave;

    const f32x4* xr = (const f32x4*)(x + (size_t)row * DIMS) + lane * 2;
    f32x4 a = xr[0];
    f32x4 b = xr[1];

    float ss = a.x*a.x + a.y*a.y + a.z*a.z + a.w*a.w
             + b.x*b.x + b.y*b.y + b.z*b.z + b.w*b.w;
#pragma unroll
    for (int m = 16; m >= 1; m >>= 1) ss += __shfl_xor(ss, m, 32);

    const float s = 1.0f / fmaxf(sqrtf(ss), 1e-12f);

    float v[8] = { a.x*s, a.y*s, a.z*s, a.w*s, b.x*s, b.y*s, b.z*s, b.w*s };
    unsigned short hi[8], lo[8];
#pragma unroll
    for (int i = 0; i < 8; ++i) {
        hi[i] = f2bf(v[i]);
        lo[i] = f2bf(v[i] - bf2f(hi[i]));
    }

    u32x4 ph, pl;
    ph.x = (unsigned)hi[0] | ((unsigned)hi[1] << 16);
    ph.y = (unsigned)hi[2] | ((unsigned)hi[3] << 16);
    ph.z = (unsigned)hi[4] | ((unsigned)hi[5] << 16);
    ph.w = (unsigned)hi[6] | ((unsigned)hi[7] << 16);
    pl.x = (unsigned)lo[0] | ((unsigned)lo[1] << 16);
    pl.y = (unsigned)lo[2] | ((unsigned)lo[3] << 16);
    pl.z = (unsigned)lo[4] | ((unsigned)lo[5] << 16);
    pl.w = (unsigned)lo[6] | ((unsigned)lo[7] << 16);

    ((u32x4*)(ghi + (size_t)row * DIMS))[lane] = ph;
    ((u32x4*)(glo + (size_t)row * DIMS))[lane] = pl;
}

// ---------------------------------------------------------------------------
// Kernel 2: fused Gram + row-max via V_WMMA_F32_16X16X32_BF16 (hi/lo split).
// Block = 256 threads = 8 waves; each wave owns one 16-row tile (A in regs),
// all 8 waves share each 16-col B tile through LDS. B tiles are streamed
// with double-buffered GLOBAL_LOAD_ASYNC_TO_LDS_B128 (ASYNCcnt) so the DMA
// of tile t+1 overlaps the 24 WMMAs on tile t. Grid splits the column range
// in 2 halves -> partial row-max per half (merged in kernel 3).
// ---------------------------------------------------------------------------
__global__ void __launch_bounds__(256) koleo_gram_rowmax(
    const unsigned short* __restrict__ ghi,
    const unsigned short* __restrict__ glo,
    float* __restrict__ rowmax)
{
    // Two 16KB buffers: per buffer [0..511]=hi tile (16x256 bf16), [512..1023]=lo.
    __shared__ u32x4 sB[2][1024];

    const int wave  = threadIdx.x >> 5;
    const int lane  = threadIdx.x & 31;
    const int half  = blockIdx.x & 1;
    const int rblk  = blockIdx.x >> 1;
    const int i0    = rblk * 128 + wave * 16;
    const int hiGrp = lane >> 4;                 // 0: lanes 0-15, 1: lanes 16-31
    const int arow  = i0 + (lane & 15);          // A: lanes 0-15 hold M=0..15 (dup)

    const unsigned tid     = threadIdx.x;
    const unsigned ldsBase = (unsigned)(size_t)&sB[0][0];  // LDS aperture low bits

    // --- A fragments (16 rows x 256, hi+lo) pinned in registers -----------
    // ISA A layout (16-bit, 16x32): lane group holds K=[klo..klo+7] and
    // K=[klo+16..klo+23], klo = k0 + 8*hiGrp  -> two 16B chunks per frag.
    u32x4 Ah[8][2], Al[8][2];
    {
        const unsigned short* ah = ghi + (size_t)arow * DIMS;
        const unsigned short* al = glo + (size_t)arow * DIMS;
#pragma unroll
        for (int ks = 0; ks < 8; ++ks) {
            const int klo = ks * 32 + hiGrp * 8;
            Ah[ks][0] = *(const u32x4*)(ah + klo);
            Ah[ks][1] = *(const u32x4*)(ah + klo + 16);
            Al[ks][0] = *(const u32x4*)(al + klo);
            Al[ks][1] = *(const u32x4*)(al + klo + 16);
        }
    }

    float vmax[8];
#pragma unroll
    for (int r = 0; r < 8; ++r) vmax[r] = -1e30f;

    // C layout: lane l, VGPR r -> M = r + 8*hiGrp, N = l&15. Diagonal element
    // of a diagonal tile sits at r == dr where dr = (l&15) - 8*hiGrp in [0,8).
    const int  dr      = (int)(lane & 15) - hiGrp * 8;
    const bool drValid = (dr >= 0) && (dr < 8);

    const int jtBegin = half * 512;
    const int jtEnd   = jtBegin + 512;

    // Issue async DMA of B tile jt into LDS buffer `buf` (16KB: 8KB hi + 8KB lo).
    auto issueTile = [&](int jt, int buf) {
        const unsigned offH = (unsigned)jt * 8192u + tid * 16u;   // bytes in ghi/glo
        const unsigned lb   = ldsBase + (unsigned)buf * 16384u + tid * 16u;
        async_ld16(lb,          ghi, offH);          // hi, first 4KB chunk
        async_ld16(lb + 4096u,  ghi, offH + 4096u);  // hi, second 4KB chunk
        async_ld16(lb + 8192u,  glo, offH);          // lo, first 4KB chunk
        async_ld16(lb + 12288u, glo, offH + 4096u);  // lo, second 4KB chunk
    };

    issueTile(jtBegin, 0);   // prologue: prime buffer 0

    for (int jt = jtBegin; jt < jtEnd; ++jt) {
        const int buf = (jt - jtBegin) & 1;

        wait_async0();       // this wave's DMA into `buf` is complete
        __syncthreads();     // ...and every other wave's too; buf^1 is dead

        if (jt + 1 < jtEnd) issueTile(jt + 1, buf ^ 1);   // overlap next DMA

        const int j0 = jt * 16;

        // 3 independent WMMA accumulator chains for ILP:
        // g = Ahi*Bhi + Ahi*Blo + Alo*Bhi   (lo*lo dropped, ~2^-18 rel.)
        v8f c0 = {}, c1 = {}, c2 = {};
        const int jrow    = lane & 15;           // B: lane holds column N = l&15
        const int bufBase = buf << 10;           // u32x4 index of buffer start
        const u32x4* sBq  = &sB[0][0];
#pragma unroll
        for (int ks = 0; ks < 8; ++ks) {
            const int kb = ks * 32 + hiGrp * 8;  // B: K=[kb..kb+7] and [kb+16..kb+23]
            const int qi = bufBase + (jrow << 5) + (kb >> 3);
            Frag ah, al, bh, bl;
            ah.q[0] = Ah[ks][0]; ah.q[1] = Ah[ks][1];
            al.q[0] = Al[ks][0]; al.q[1] = Al[ks][1];
            bh.q[0] = sBq[qi];        bh.q[1] = sBq[qi + 2];
            bl.q[0] = sBq[qi + 512];  bl.q[1] = sBq[qi + 514];

            c0 = __builtin_amdgcn_wmma_f32_16x16x32_bf16(
                     false, ah.v, false, bh.v, (short)0, c0, false, false);
            c1 = __builtin_amdgcn_wmma_f32_16x16x32_bf16(
                     false, ah.v, false, bl.v, (short)0, c1, false, false);
            c2 = __builtin_amdgcn_wmma_f32_16x16x32_bf16(
                     false, al.v, false, bh.v, (short)0, c2, false, false);
        }

        const bool diag = (j0 == i0);
#pragma unroll
        for (int r = 0; r < 8; ++r) {
            const float g = c0[r] + c1[r] + c2[r];
            if (diag && drValid && r == dr) continue;   // skip <xi,xi>
            vmax[r] = fmaxf(vmax[r], g);
        }
    }

    // Reduce each row's max across the 16 lanes of its lane group.
#pragma unroll
    for (int r = 0; r < 8; ++r) {
#pragma unroll
        for (int m = 8; m >= 1; m >>= 1)
            vmax[r] = fmaxf(vmax[r], __shfl_xor(vmax[r], m, 32));
    }
    float* outp = rowmax + (size_t)half * N_ROWS;
    if (lane == 0) {
#pragma unroll
        for (int r = 0; r < 8; ++r) outp[i0 + r] = vmax[r];
    } else if (lane == 16) {
#pragma unroll
        for (int r = 0; r < 8; ++r) outp[i0 + 8 + r] = vmax[r];
    }
}

// ---------------------------------------------------------------------------
// Kernel 3: deterministic final reduction -> scalar loss.
// ---------------------------------------------------------------------------
__global__ void __launch_bounds__(256) koleo_reduce(
    const float* __restrict__ rowmax, float* __restrict__ out)
{
    __shared__ float sm[256];
    float s = 0.0f;
    for (int i = threadIdx.x; i < N_ROWS; i += 256) {
        const float g  = fmaxf(rowmax[i], rowmax[i + N_ROWS]);
        const float sq = fmaxf(2.0f - 2.0f * g, 0.0f);
        s += logf(sqrtf(sq) + 1e-8f);
    }
    sm[threadIdx.x] = s;
    __syncthreads();
    for (int w = 128; w > 0; w >>= 1) {
        if (threadIdx.x < w) sm[threadIdx.x] += sm[threadIdx.x + w];
        __syncthreads();
    }
    if (threadIdx.x == 0) out[0] = -sm[0] / (float)N_ROWS;
}

// ---------------------------------------------------------------------------
extern "C" void kernel_launch(void* const* d_in, const int* in_sizes, int n_in,
                              void* d_out, int out_size, void* d_ws, size_t ws_size,
                              hipStream_t stream)
{
    (void)in_sizes; (void)n_in; (void)out_size; (void)ws_size;
    const float* x = (const float*)d_in[0];

    char* ws = (char*)d_ws;
    unsigned short* ghi    = (unsigned short*)ws;                                   // 8 MB
    unsigned short* glo    = (unsigned short*)(ws + (size_t)N_ROWS * DIMS * 2);     // 8 MB
    float*          rowmax = (float*)        (ws + (size_t)N_ROWS * DIMS * 4);      // 128 KB

    koleo_normalize_split<<<N_ROWS / 8, 256, 0, stream>>>(x, ghi, glo);
    koleo_gram_rowmax<<<(N_ROWS / 128) * 2, 256, 0, stream>>>(ghi, glo, rowmax);
    koleo_reduce<<<1, 256, 0, stream>>>(rowmax, (float*)d_out);
}
